// HeteroGNN_30193620091084
// MI455X (gfx1250) — matrix operand order repaired and checked by reading, not compiled, not run on hardware
//
#include <hip/hip_runtime.h>
#include <hip/hip_bf16.h>

// HeteroGNN (bipartite 2-layer SAGE) for MI455X / gfx1250.
//
// Sparse scatter (1M edges x 128/64/128 floats of L2-resident atomics)
// dominates; dense transforms (~16 GFLOP) run on the WMMA pipe as
// f16 x f16 -> f32 (v_wmma_f32_16x16x32_f16). Weights are pre-swizzled to
// f16 in the exact B-fragment VGPR layout so each fragment is one 32-byte
// coalesced load; segment-mean is pre-applied so A fragments are 4x b128
// loads + packed converts. wave32: one wave per 16-row x N output tile.

typedef __attribute__((ext_vector_type(16))) _Float16 v16h;
typedef __attribute__((ext_vector_type(8)))  float    v8f;

// ---------------------------------------------------------------------------
// Scatter: agg[dst] += x[src], deg[dst] += 1   (one thread per edge x 4 cols)
// ---------------------------------------------------------------------------
__global__ void k_scatter_add(const float* __restrict__ xsrc,
                              const int* __restrict__ src,
                              const int* __restrict__ dst,
                              float* __restrict__ agg,
                              float* __restrict__ deg,
                              int E, int D) {
  int t = blockIdx.x * blockDim.x + threadIdx.x;
  int chunks = D >> 2;
  int total = E * chunks;
  if (t >= total) return;
  int e = t / chunks;
  int c = t - e * chunks;
  int s = src[e];
  int d = dst[e];
  const float4 v = *(const float4*)(xsrc + (size_t)s * D + (size_t)c * 4);
  float* ap = agg + (size_t)d * D + (size_t)c * 4;
  atomicAdd(ap + 0, v.x);
  atomicAdd(ap + 1, v.y);
  atomicAdd(ap + 2, v.z);
  atomicAdd(ap + 3, v.w);
  if (deg != nullptr && c == 0) atomicAdd(deg + d, 1.0f);
}

// ---------------------------------------------------------------------------
// Mean: agg[row,:] *= 1/max(deg[row],1)   (in place, float4 per thread)
// ---------------------------------------------------------------------------
__global__ void k_mean(float* __restrict__ agg, const float* __restrict__ deg,
                       int Nn, int D) {
  int t = blockIdx.x * blockDim.x + threadIdx.x;
  int chunks = D >> 2;
  int total = Nn * chunks;
  if (t >= total) return;
  int row = t / chunks;
  float s = 1.0f / fmaxf(deg[row], 1.0f);
  float4* p = (float4*)(agg + (size_t)t * 4);
  float4 v = *p;
  v.x *= s; v.y *= s; v.z *= s; v.w *= s;
  *p = v;
}

// ---------------------------------------------------------------------------
// Weight swizzle: [Wl;Wr] (Ktot x N, f32) -> f16 B-fragment order:
//   swz[ ((ks*NT + ct)*32 + lane)*16 + j ] = W[ks*32 + (lane>=16?8:0) + (j<8?j:j+8)][ct*16 + (lane&15)]
// so a wave's B fragment for (ks,ct) is 32 contiguous bytes per lane.
// ---------------------------------------------------------------------------
__global__ void k_swizzle_w(const float* __restrict__ Wl,
                            const float* __restrict__ Wr,
                            int Ka, int Kb, int N, int NT_,
                            _Float16* __restrict__ swz) {
  int idx = blockIdx.x * blockDim.x + threadIdx.x;
  int total = (Ka + Kb) * N;
  if (idx >= total) return;
  int j    = idx & 15;
  int lane = (idx >> 4) & 31;
  int tile = idx >> 9;            // ks*NT + ct
  int ct = tile % NT_;
  int ks = tile / NT_;
  int kg = ks * 32 + ((lane >= 16) ? 8 : 0) + ((j < 8) ? j : j + 8);
  int col = ct * 16 + (lane & 15);
  float v = (kg < Ka) ? Wl[(size_t)kg * N + col]
                      : Wr[(size_t)(kg - Ka) * N + col];
  swz[idx] = (_Float16)v;
}

// ---------------------------------------------------------------------------
// Fused SAGE GEMM: out[m,:] = act( [Aagg | Ax][m,:] @ swzW + bias )
// (Aagg already mean-scaled.) Ka may be 0. NT = N/16 column tiles.
// One wave per 16-row tile, 8 waves per block.
// ---------------------------------------------------------------------------
template <int NT>
__global__ __launch_bounds__(256)
void k_sage_gemm(const float* Aagg, int Ka,
                 const float* __restrict__ Ax, int Kb,
                 const _Float16* __restrict__ swzW,
                 const float* __restrict__ bias,
                 float* out, int M, int N, int relu) {
  const int lane = threadIdx.x & 31;
  const int wave = threadIdx.x >> 5;
  const int rowTile = blockIdx.x * 8 + wave;     // wave-uniform
  if (rowTile * 16 >= M) return;
  const int rowBase = rowTile * 16;
  const int row = rowBase + (lane & 15);
  const int hi = lane >> 4;
  const int b0 = hi ? 8 : 0;                     // A K-run starts (ISA 7.12.2)
  const int b1 = hi ? 24 : 16;

  v8f acc[NT];
#pragma unroll
  for (int i = 0; i < NT; ++i)
    acc[i] = (v8f){0.f, 0.f, 0.f, 0.f, 0.f, 0.f, 0.f, 0.f};

  const int Ktot = Ka + Kb;
  for (int kg = 0; kg < Ktot; kg += 32) {
    // ---- A fragment: two contiguous 8-float runs -> 4x b128 + packed cvt ----
    const float* abase = (kg < Ka)
        ? (Aagg + (size_t)row * Ka + kg)
        : (Ax   + (size_t)row * Kb + (kg - Ka));
    const float4 f0 = *(const float4*)(abase + b0);
    const float4 f1 = *(const float4*)(abase + b0 + 4);
    const float4 f2 = *(const float4*)(abase + b1);
    const float4 f3 = *(const float4*)(abase + b1 + 4);
    v16h a;
    a[0]  = (_Float16)f0.x; a[1]  = (_Float16)f0.y;
    a[2]  = (_Float16)f0.z; a[3]  = (_Float16)f0.w;
    a[4]  = (_Float16)f1.x; a[5]  = (_Float16)f1.y;
    a[6]  = (_Float16)f1.z; a[7]  = (_Float16)f1.w;
    a[8]  = (_Float16)f2.x; a[9]  = (_Float16)f2.y;
    a[10] = (_Float16)f2.z; a[11] = (_Float16)f2.w;
    a[12] = (_Float16)f3.x; a[13] = (_Float16)f3.y;
    a[14] = (_Float16)f3.z; a[15] = (_Float16)f3.w;

    // ---- B fragments: one 32-byte contiguous v16h load per column tile ----
    const int ks = kg >> 5;
    const _Float16* wbase = swzW + (((size_t)ks * NT) * 32 + lane) * 16;
#pragma unroll
    for (int ct = 0; ct < NT; ++ct) {
      const v16h b = *(const v16h*)(wbase + (size_t)ct * 32 * 16);
      acc[ct] = __builtin_amdgcn_wmma_f32_16x16x32_f16(
          /*neg_a=*/false, a, /*neg_b=*/false, b,
          /*c_mod=*/(short)0, acc[ct], /*reuse_a=*/false, /*reuse_b=*/false);
    }
  }

  // ---- epilogue: bias, optional ReLU, store ----
#pragma unroll
  for (int ct = 0; ct < NT; ++ct) {
    const int col = ct * 16 + (lane & 15);
    const float bb = bias[col];
#pragma unroll
    for (int v = 0; v < 8; ++v) {
      float f = acc[ct][v] + bb;
      if (relu) f = fmaxf(f, 0.0f);
      out[(size_t)(rowBase + v + (hi ? 8 : 0)) * N + col] = f;
    }
  }
}

// ---------------------------------------------------------------------------
// Launch
// ---------------------------------------------------------------------------
extern "C" void kernel_launch(void* const* d_in, const int* in_sizes, int n_in,
                              void* d_out, int out_size, void* d_ws, size_t ws_size,
                              hipStream_t stream) {
  const float* x_c     = (const float*)d_in[0];   // [NC,128]
  const float* x_m     = (const float*)d_in[1];   // [NM,64]
  const float* W1cm_l  = (const float*)d_in[2];   // [128,128]
  const float* W1cm_r  = (const float*)d_in[3];   // [64,128]
  const float* b1_cm   = (const float*)d_in[4];   // [128]
  const float* W1mc_l  = (const float*)d_in[5];   // [64,128]
  const float* W1mc_r  = (const float*)d_in[6];   // [128,128]
  const float* b1_mc   = (const float*)d_in[7];   // [128]
  // d_in[8..10]: W2cm_l, W2cm_r, b2_cm -- dead in reference output
  const float* W2mc_l  = (const float*)d_in[11];  // [128,128]
  const float* W2mc_r  = (const float*)d_in[12];  // [128,128]
  const float* b2_mc   = (const float*)d_in[13];  // [128]
  const float* Wout    = (const float*)d_in[14];  // [128,64]
  const float* bout    = (const float*)d_in[15];  // [64]
  const int* ei_cm_src = (const int*)d_in[16];
  const int* ei_cm_dst = (const int*)d_in[17];
  const int* ei_mc_src = (const int*)d_in[18];
  const int* ei_mc_dst = (const int*)d_in[19];

  const int IN_C = 128, IN_M = 64, HID = 128, OUTD = 64;
  const int NC = in_sizes[0] / IN_C;
  const int NM = in_sizes[1] / IN_M;
  const int E  = in_sizes[16];

  // ---- workspace layout (64B-aligned chunks) ----
  char* base = (char*)d_ws;
  size_t off = 0;
  auto alloc = [&](size_t bytes) -> char* {
    char* p = base + off;
    off += (bytes + 63) & ~(size_t)63;
    return p;
  };
  float* deg_m = (float*)alloc((size_t)NM * 4);
  float* deg_c = (float*)alloc((size_t)NC * 4);
  size_t r1a = (size_t)NM * HID + (size_t)NC * IN_M;
  size_t r1b = (size_t)NC * HID;
  size_t r1sz = (r1a > r1b) ? r1a : r1b;
  float* R1 = (float*)alloc(r1sz * 4);
  float* agg_m  = R1;                           // [NM,128]
  float* agg_c1 = R1 + (size_t)NM * HID;        // [NC,64]
  float* agg_c2 = R1;                           // [NC,128] (reuse after A/B GEMMs)
  float* h_m = (float*)alloc((size_t)NM * HID * 4);
  float* h_c = (float*)alloc((size_t)NC * HID * 4);
  _Float16* swz1 = (_Float16*)alloc((size_t)(IN_C + IN_M) * HID * 2);
  _Float16* swz2 = (_Float16*)alloc((size_t)(IN_M + IN_C) * HID * 2);
  _Float16* swz3 = (_Float16*)alloc((size_t)(HID + HID) * HID * 2);
  _Float16* swz4 = (_Float16*)alloc((size_t)HID * OUTD * 2);

  // ---- pre-swizzle weights to f16 fragment layout (tiny, L2-hot) ----
  {
    int t1 = (IN_C + IN_M) * HID;
    k_swizzle_w<<<(t1 + 255) / 256, 256, 0, stream>>>(W1cm_l, W1cm_r, IN_C, IN_M, HID, 8, swz1);
    int t2 = (IN_M + IN_C) * HID;
    k_swizzle_w<<<(t2 + 255) / 256, 256, 0, stream>>>(W1mc_l, W1mc_r, IN_M, IN_C, HID, 8, swz2);
    int t3 = (HID + HID) * HID;
    k_swizzle_w<<<(t3 + 255) / 256, 256, 0, stream>>>(W2mc_l, W2mc_r, HID, HID, HID, 8, swz3);
    int t4 = HID * OUTD;
    k_swizzle_w<<<(t4 + 255) / 256, 256, 0, stream>>>(nullptr, Wout, 0, HID, OUTD, 4, swz4);
  }

  // ---- degrees + layer-1 aggregations ----
  hipMemsetAsync(deg_m,  0, (size_t)NM * 4, stream);
  hipMemsetAsync(deg_c,  0, (size_t)NC * 4, stream);
  hipMemsetAsync(agg_m,  0, (size_t)NM * HID  * 4, stream);
  hipMemsetAsync(agg_c1, 0, (size_t)NC * IN_M * 4, stream);

  {
    int total = E * (IN_C / 4);
    k_scatter_add<<<(total + 255) / 256, 256, 0, stream>>>(
        x_c, ei_cm_src, ei_cm_dst, agg_m, deg_m, E, IN_C);
  }
  {
    int total = E * (IN_M / 4);
    k_scatter_add<<<(total + 255) / 256, 256, 0, stream>>>(
        x_m, ei_mc_src, ei_mc_dst, agg_c1, deg_c, E, IN_M);
  }
  {
    int total = NM * (HID / 4);
    k_mean<<<(total + 255) / 256, 256, 0, stream>>>(agg_m, deg_m, NM, HID);
  }
  {
    int total = NC * (IN_M / 4);
    k_mean<<<(total + 255) / 256, 256, 0, stream>>>(agg_c1, deg_c, NC, IN_M);
  }

  // ---- layer-1 fused SAGE GEMMs (ReLU) ----
  {
    int blocks = (((NM + 15) / 16) + 7) / 8;
    k_sage_gemm<8><<<blocks, 256, 0, stream>>>(
        agg_m, IN_C, x_m, IN_M, swz1, b1_cm, h_m, NM, HID, /*relu=*/1);
  }
  {
    int blocks = (((NC + 15) / 16) + 7) / 8;
    k_sage_gemm<8><<<blocks, 256, 0, stream>>>(
        agg_c1, IN_M, x_c, IN_C, swz2, b1_mc, h_c, NC, HID, /*relu=*/1);
  }

  // ---- layer-2 aggregation of h_m over mc edges (deg_c reused) ----
  hipMemsetAsync(agg_c2, 0, (size_t)NC * HID * 4, stream);
  {
    int total = E * (HID / 4);
    k_scatter_add<<<(total + 255) / 256, 256, 0, stream>>>(
        h_m, ei_mc_src, ei_mc_dst, agg_c2, nullptr, E, HID);
  }
  {
    int total = NC * (HID / 4);
    k_mean<<<(total + 255) / 256, 256, 0, stream>>>(agg_c2, deg_c, NC, HID);
  }

  // ---- layer-2 SAGE GEMM (no ReLU), h_c2 written in place over agg_c2 ----
  {
    int blocks = (((NC + 15) / 16) + 7) / 8;
    k_sage_gemm<8><<<blocks, 256, 0, stream>>>(
        agg_c2, HID, h_c, HID, swz3, b2_mc, agg_c2, NC, HID, /*relu=*/0);
  }

  // ---- output projection: out = h_c2 @ Wout + bout ----
  {
    int blocks = (((NC + 15) / 16) + 7) / 8;
    k_sage_gemm<4><<<blocks, 256, 0, stream>>>(
        nullptr, 0, agg_c2, HID, swz4, bout, (float*)d_out, NC, OUTD, /*relu=*/0);
  }
}
